// W8A8LinearStatic_39204461478515
// MI455X (gfx1250) — 8-bit matrix operand order repaired and checked by reading, compiled, not run on hardware
//
#include <hip/hip_runtime.h>
#include <hip/hip_bf16.h>

typedef __attribute__((ext_vector_type(8))) int v8i;
typedef __attribute__((ext_vector_type(4))) int v4i;
typedef __attribute__((ext_vector_type(2))) int v2i;

#define M_DIM 4096
#define K_DIM 4096
#define N_DIM 11008

__global__ __launch_bounds__(256) void quant_x_kernel(
    const float* __restrict__ x, const float* __restrict__ input_scale,
    const int* __restrict__ input_offset, unsigned* __restrict__ xq, long n4) {
  long i = (long)blockIdx.x * blockDim.x + threadIdx.x;
  if (i >= n4) return;
  float inv = 1.0f / input_scale[0];
  int off = input_offset[0];
  const float4* xp = (const float4*)x;
  float4 v = xp[i];
  int a0 = (int)rintf(v.x * inv) + off;
  int a1 = (int)rintf(v.y * inv) + off;
  int a2 = (int)rintf(v.z * inv) + off;
  int a3 = (int)rintf(v.w * inv) + off;
  a0 = a0 < -128 ? -128 : (a0 > 127 ? 127 : a0);
  a1 = a1 < -128 ? -128 : (a1 > 127 ? 127 : a1);
  a2 = a2 < -128 ? -128 : (a2 > 127 ? 127 : a2);
  a3 = a3 < -128 ? -128 : (a3 > 127 ? 127 : a3);
  xq[i] = (unsigned)(a0 & 255) | ((unsigned)(a1 & 255) << 8) |
          ((unsigned)(a2 & 255) << 16) | ((unsigned)(a3 & 255) << 24);
}

__global__ __launch_bounds__(256) void pack_w_kernel(
    const int* __restrict__ w, unsigned* __restrict__ wq, long n4) {
  long i = (long)blockIdx.x * blockDim.x + threadIdx.x;
  if (i >= n4) return;
  const int4* wp = (const int4*)w;
  int4 v = wp[i];
  wq[i] = (unsigned)(v.x & 255) | ((unsigned)(v.y & 255) << 8) |
          ((unsigned)(v.z & 255) << 16) | ((unsigned)(v.w & 255) << 24);
}

__global__ __launch_bounds__(256) void gemm_i8_wmma_kernel(
    const char* __restrict__ Aq, const char* __restrict__ Wq,
    const int* __restrict__ qbias, const float* __restrict__ dscale,
    float* __restrict__ out) {
  const int tid = threadIdx.x;
  const int wave = tid >> 5;
  const int lane = tid & 31;
  const int lane16 = lane & 15;
  const int laneHi = lane >> 4;
  const int waveM = wave >> 2;
  const int waveN = wave & 3;

  const long m_base = (long)blockIdx.y * 128 + (long)waveM * 64;
  const long n_base = (long)blockIdx.x * 256 + (long)waveN * 64;

  const char* arow[4];
  const char* brow[4];
  {
    const char* a0 = Aq + (m_base + lane16) * (long)K_DIM + laneHi * 8;
    const char* b0 = Wq + (n_base + lane16) * (long)K_DIM + laneHi * 8;
#pragma unroll
    for (int i = 0; i < 4; ++i) { arow[i] = a0 + (long)i * 16 * K_DIM; }
#pragma unroll
    for (int j = 0; j < 4; ++j) { brow[j] = b0 + (long)j * 16 * K_DIM; }
  }

  v8i acc[4][4];
#pragma unroll
  for (int i = 0; i < 4; ++i)
#pragma unroll
    for (int j = 0; j < 4; ++j) acc[i][j] = (v8i)0;

  for (int kc = 0; kc < K_DIM; kc += 64) {
    v8i a[4], b[4];
#pragma unroll
    for (int i = 0; i < 4; ++i) {
      v2i t0 = *(const v2i*)(arow[i] + kc);
      v2i t1 = *(const v2i*)(arow[i] + kc + 16);
      v2i t2 = *(const v2i*)(arow[i] + kc + 32);
      v2i t3 = *(const v2i*)(arow[i] + kc + 48);
      a[i][0] = t0[0]; a[i][1] = t0[1]; a[i][2] = t1[0]; a[i][3] = t1[1];
      a[i][4] = t2[0]; a[i][5] = t2[1]; a[i][6] = t3[0]; a[i][7] = t3[1];
    }
#pragma unroll
    for (int j = 0; j < 4; ++j) {
      v2i t0 = *(const v2i*)(brow[j] + kc);
      v2i t1 = *(const v2i*)(brow[j] + kc + 16);
      v2i t2 = *(const v2i*)(brow[j] + kc + 32);
      v2i t3 = *(const v2i*)(brow[j] + kc + 48);
      b[j][0] = t0[0]; b[j][1] = t0[1]; b[j][2] = t1[0]; b[j][3] = t1[1];
      b[j][4] = t2[0]; b[j][5] = t2[1]; b[j][6] = t3[0]; b[j][7] = t3[1];
    }

#pragma unroll
    for (int i = 0; i < 4; ++i)
#pragma unroll
      for (int j = 0; j < 4; ++j)
        acc[i][j] = __builtin_amdgcn_wmma_i32_16x16x64_iu8(
            true, a[i], true, b[j], acc[i][j], false, false);
  }

  {
    const long m0 = m_base + laneHi * 8;
    const long n0 = n_base + lane16;
#pragma unroll
    for (int tj = 0; tj < 4; ++tj) {
      const long n = n0 + tj * 16;
      const int   qb = qbias[n];
      const float sc = dscale[n];
#pragma unroll
      for (int ti = 0; ti < 4; ++ti) {
        float* op = out + (m0 + ti * 16) * (long)N_DIM + n;
#pragma unroll
        for (int j = 0; j < 8; ++j)
          op[j * (long)N_DIM] = (float)(acc[ti][tj][j] + qb) * sc;
      }
    }
  }
}

extern "C" void kernel_launch(void* const* d_in, const int* in_sizes, int n_in,
                              void* d_out, int out_size, void* d_ws, size_t ws_size,
                              hipStream_t stream) {
  const float* x           = (const float*)d_in[0];
  const int*   weight      = (const int*)d_in[1];
  const float* deq_scale   = (const float*)d_in[2];
  const float* input_scale = (const float*)d_in[3];
  const int*   input_offset= (const int*)d_in[4];
  const int*   quant_bias  = (const int*)d_in[5];
  float*       out         = (float*)d_out;

  char* ws = (char*)d_ws;
  unsigned* xq = (unsigned*)ws;
  unsigned* wq = (unsigned*)(ws + (size_t)M_DIM * K_DIM);

  long xn4 = (long)M_DIM * K_DIM / 4;
  quant_x_kernel<<<(unsigned)((xn4 + 255) / 256), 256, 0, stream>>>(
      x, input_scale, input_offset, xq, xn4);

  long wn4 = (long)N_DIM * K_DIM / 4;
  pack_w_kernel<<<(unsigned)((wn4 + 255) / 256), 256, 0, stream>>>(
      weight, wq, wn4);

  dim3 grid(N_DIM / 256, M_DIM / 128);
  gemm_i8_wmma_kernel<<<grid, 256, 0, stream>>>(
      (const char*)xq, (const char*)wq, quant_bias, deq_scale, out);
}
